// GCN_82197084110895
// MI455X (gfx1250) — matrix operand order repaired and checked by reading, not compiled, not run on hardware
//
#include <hip/hip_runtime.h>
#include <hip/hip_bf16.h>

typedef __attribute__((ext_vector_type(16))) _Float16 v16h;
typedef __attribute__((ext_vector_type(8)))  float    v8f;

// ---------------------------------------------------------------------------
// 1) Degree counting: deg_out[src[e]] += 1, deg_in[dst[e]] += 1
// ---------------------------------------------------------------------------
__global__ void gcn_degrees(const int* __restrict__ src, const int* __restrict__ dst,
                            float* __restrict__ deg_out, float* __restrict__ deg_in, int E) {
    int i = blockIdx.x * blockDim.x + threadIdx.x;
    if (i >= E) return;
    unsafeAtomicAdd(deg_out + src[i], 1.0f);
    unsafeAtomicAdd(deg_in  + dst[i], 1.0f);
}

// ---------------------------------------------------------------------------
// 2) s = rsqrt(max(deg, 1))  (DGL clamps zero degree to 1)
// ---------------------------------------------------------------------------
__global__ void gcn_rsqrt(const float* __restrict__ deg_out, const float* __restrict__ deg_in,
                          float* __restrict__ s_out, float* __restrict__ s_in, int N) {
    int i = blockIdx.x * blockDim.x + threadIdx.x;
    if (i >= N) return;
    s_out[i] = rsqrtf(fmaxf(deg_out[i], 1.0f));
    s_in[i]  = rsqrtf(fmaxf(deg_in[i],  1.0f));
}

// ---------------------------------------------------------------------------
// 3) Pack W (row-major f32, 64 x ncols) into per-lane B-matrix layout (f16).
//    Index: (((cb*2 + ks)*32 + lane)*16 + j) halves. lane lo = column cb*16+lo,
//    lane hi selects K 0..15 vs 16..31 within the 32-wide K step, j = K in half.
//    Columns >= ncols zero-padded (40 -> 48 pad for layer 1).
// ---------------------------------------------------------------------------
__global__ void gcn_packW(const float* __restrict__ W, _Float16* __restrict__ Wp,
                          int ncols, int total) {
    int idx = blockIdx.x * blockDim.x + threadIdx.x;
    if (idx >= total) return;
    int j    =  idx        & 15;
    int lane = (idx >> 4)  & 31;
    int ks   = (idx >> 9)  & 1;
    int cb   =  idx >> 10;
    int lo = lane & 15, hi = lane >> 4;
    int row = ks * 32 + hi * 16 + j;   // K index in [0,64)
    int col = cb * 16 + lo;
    float v = (col < ncols) ? W[row * ncols + col] : 0.0f;
    Wp[idx] = (_Float16)v;
}

// ---------------------------------------------------------------------------
// 4) WMMA GEMM:  H[row, :] = (X[row, :] * (S ? S[row] : 1)) @ W   (K = 64)
//    One wave = 16 rows x COLT*16 cols. COLT is a template constant so the
//    column loop fully unrolls: constant-offset b128 B loads + COLT*2 WMMAs.
//    Stores: wave-uniform fast path (no EXEC churn) vs guarded tail path.
// ---------------------------------------------------------------------------
template <int COLT>
__global__ void gcn_gemm_wmma(const float* __restrict__ X,
                              const float* __restrict__ S,
                              const _Float16* __restrict__ Wp,
                              float* __restrict__ H,
                              int nrows, int strideOut) {
    const int lane = threadIdx.x & 31;
    const int wave = threadIdx.x >> 5;
    const int wavesPerBlock = blockDim.x >> 5;
    const int tile = blockIdx.x * wavesPerBlock + wave;
    const int rowBase = tile << 4;
    if (rowBase >= nrows) return;          // wave-uniform: EXEC stays all-ones

    const int lo = lane & 15;
    const int hi = lane >> 4;

    int row = rowBase + lo;
    if (row >= nrows) row = nrows - 1;     // clamp (tail stores are guarded)
    const float s = S ? S[row] : 1.0f;

    const float4* xr = (const float4*)(X + (size_t)row * 64);

    // B tiles up-front: contiguous 32B per lane -> two b128 loads each, all clause-able
    v16h b[COLT * 2];
#pragma unroll
    for (int t = 0; t < COLT * 2; ++t)
        b[t] = *(const v16h*)(Wp + ((size_t)t * 32 + lane) * 16);

    // A tiles (contiguous K order, same as packed B): lane-half selects K 0-15 vs 16-31
    v16h a0, a1;
    {
        float4 q0 = xr[hi * 4 + 0];
        float4 q1 = xr[hi * 4 + 1];
        float4 q2 = xr[hi * 4 + 2];
        float4 q3 = xr[hi * 4 + 3];
        a0[0]=(_Float16)(q0.x*s); a0[1]=(_Float16)(q0.y*s); a0[2]=(_Float16)(q0.z*s); a0[3]=(_Float16)(q0.w*s);
        a0[4]=(_Float16)(q1.x*s); a0[5]=(_Float16)(q1.y*s); a0[6]=(_Float16)(q1.z*s); a0[7]=(_Float16)(q1.w*s);
        a0[8]=(_Float16)(q2.x*s); a0[9]=(_Float16)(q2.y*s); a0[10]=(_Float16)(q2.z*s); a0[11]=(_Float16)(q2.w*s);
        a0[12]=(_Float16)(q3.x*s); a0[13]=(_Float16)(q3.y*s); a0[14]=(_Float16)(q3.z*s); a0[15]=(_Float16)(q3.w*s);

        float4 r0 = xr[8  + hi * 4 + 0];
        float4 r1 = xr[8  + hi * 4 + 1];
        float4 r2 = xr[8  + hi * 4 + 2];
        float4 r3 = xr[8  + hi * 4 + 3];
        a1[0]=(_Float16)(r0.x*s); a1[1]=(_Float16)(r0.y*s); a1[2]=(_Float16)(r0.z*s); a1[3]=(_Float16)(r0.w*s);
        a1[4]=(_Float16)(r1.x*s); a1[5]=(_Float16)(r1.y*s); a1[6]=(_Float16)(r1.z*s); a1[7]=(_Float16)(r1.w*s);
        a1[8]=(_Float16)(r2.x*s); a1[9]=(_Float16)(r2.y*s); a1[10]=(_Float16)(r2.z*s); a1[11]=(_Float16)(r2.w*s);
        a1[12]=(_Float16)(r3.x*s); a1[13]=(_Float16)(r3.y*s); a1[14]=(_Float16)(r3.z*s); a1[15]=(_Float16)(r3.w*s);
    }

    v8f c[COLT];
#pragma unroll
    for (int cb = 0; cb < COLT; ++cb) {
        c[cb] = (v8f){};
        c[cb] = __builtin_amdgcn_wmma_f32_16x16x32_f16(false, a0, false, b[cb*2+0], (short)0, c[cb], false, false);
        c[cb] = __builtin_amdgcn_wmma_f32_16x16x32_f16(false, a1, false, b[cb*2+1], (short)0, c[cb], false, false);
    }

    // C/D layout: lane-half selects M 0-7 vs 8-15; column = cb*16 + lo
    float* hb = H + (size_t)(rowBase + hi * 8) * strideOut + lo;
    if (rowBase + 16 <= nrows) {           // wave-uniform fast path: no EXEC masking
#pragma unroll
        for (int cb = 0; cb < COLT; ++cb)
#pragma unroll
            for (int r = 0; r < 8; ++r)
                hb[(size_t)r * strideOut + cb * 16] = c[cb][r];
    } else {                               // tail tile: guard each row
#pragma unroll
        for (int cb = 0; cb < COLT; ++cb)
#pragma unroll
            for (int r = 0; r < 8; ++r) {
                int rr = rowBase + hi * 8 + r;
                if (rr < nrows) H[(size_t)rr * strideOut + cb * 16 + lo] = c[cb][r];
            }
    }
}

// ---------------------------------------------------------------------------
// 5) Layer-0 scatter: agg[dst[e], :64] += H[src[e], :64]; thread = (edge, 16-feat chunk)
// ---------------------------------------------------------------------------
__global__ void gcn_scatter64(const int* __restrict__ src, const int* __restrict__ dst,
                              const float* __restrict__ H, float* __restrict__ agg, int E) {
    int idx = blockIdx.x * blockDim.x + threadIdx.x;
    if (idx >= E * 4) return;
    int e = idx >> 2, ch = idx & 3;
    int s = src[e], d = dst[e];
    const float4* hp = (const float4*)(H + (size_t)s * 64 + ch * 16);
    float* ap = agg + (size_t)d * 64 + ch * 16;
#pragma unroll
    for (int k = 0; k < 4; ++k) {
        float4 v = hp[k];
        unsafeAtomicAdd(ap + 4 * k + 0, v.x);
        unsafeAtomicAdd(ap + 4 * k + 1, v.y);
        unsafeAtomicAdd(ap + 4 * k + 2, v.z);
        unsafeAtomicAdd(ap + 4 * k + 3, v.w);
    }
}

// ---------------------------------------------------------------------------
// 6) X1 = relu(agg * s_in + b0) * s_out   (fold layer-1 out-degree scale in)
// ---------------------------------------------------------------------------
__global__ void gcn_relu_scale(const float* __restrict__ agg, const float* __restrict__ s_in,
                               const float* __restrict__ s_out, const float* __restrict__ bias,
                               float* __restrict__ X1, int N) {
    int idx = blockIdx.x * blockDim.x + threadIdx.x;
    if (idx >= N * 16) return;
    int i = idx >> 4, f4 = idx & 15;
    float si = s_in[i], so = s_out[i];
    float4 v = ((const float4*)agg)[idx];
    float4 b = ((const float4*)bias)[f4];
    float4 r;
    r.x = fmaxf(v.x * si + b.x, 0.0f) * so;
    r.y = fmaxf(v.y * si + b.y, 0.0f) * so;
    r.z = fmaxf(v.z * si + b.z, 0.0f) * so;
    r.w = fmaxf(v.w * si + b.w, 0.0f) * so;
    ((float4*)X1)[idx] = r;
}

// ---------------------------------------------------------------------------
// 7) Layer-1 scatter: out[dst[e], :40] += H2[src[e], :40]; H2 padded stride 48
// ---------------------------------------------------------------------------
__global__ void gcn_scatter40(const int* __restrict__ src, const int* __restrict__ dst,
                              const float* __restrict__ H2, float* __restrict__ out, int E) {
    int idx = blockIdx.x * blockDim.x + threadIdx.x;
    if (idx >= E * 5) return;
    int e = idx / 5;
    int ch = idx - e * 5;
    int s = src[e], d = dst[e];
    const float4* hp = (const float4*)(H2 + (size_t)s * 48 + ch * 8);
    float* ap = out + (size_t)d * 40 + ch * 8;
    float4 v0 = hp[0], v1 = hp[1];
    unsafeAtomicAdd(ap + 0, v0.x);
    unsafeAtomicAdd(ap + 1, v0.y);
    unsafeAtomicAdd(ap + 2, v0.z);
    unsafeAtomicAdd(ap + 3, v0.w);
    unsafeAtomicAdd(ap + 4, v1.x);
    unsafeAtomicAdd(ap + 5, v1.y);
    unsafeAtomicAdd(ap + 6, v1.z);
    unsafeAtomicAdd(ap + 7, v1.w);
}

// ---------------------------------------------------------------------------
// 8) Finalize in place: out = out * s_in + b1   (40 cols = 10 float4 per row)
// ---------------------------------------------------------------------------
__global__ void gcn_final(float* __restrict__ out, const float* __restrict__ s_in,
                          const float* __restrict__ bias, int N) {
    int idx = blockIdx.x * blockDim.x + threadIdx.x;
    if (idx >= N * 10) return;
    int i = idx / 10;
    int c4 = idx - i * 10;
    float si = s_in[i];
    float4 v = ((const float4*)out)[idx];
    float4 b = ((const float4*)bias)[c4];
    v.x = v.x * si + b.x;
    v.y = v.y * si + b.y;
    v.z = v.z * si + b.z;
    v.w = v.w * si + b.w;
    ((float4*)out)[idx] = v;
}

// ---------------------------------------------------------------------------
extern "C" void kernel_launch(void* const* d_in, const int* in_sizes, int n_in,
                              void* d_out, int out_size, void* d_ws, size_t ws_size,
                              hipStream_t stream) {
    const float* in_feat = (const float*)d_in[0];
    const int*   src     = (const int*)d_in[1];
    const int*   dst     = (const int*)d_in[2];
    const float* W0      = (const float*)d_in[3];
    const float* b0      = (const float*)d_in[4];
    const float* W1      = (const float*)d_in[5];
    const float* b1      = (const float*)d_in[6];
    float* out = (float*)d_out;

    const int N = in_sizes[0] / 64;   // 100000
    const int E = in_sizes[1];        // 1600000

    // Workspace (floats): deg_out | deg_in | s_out | s_in | H0/X1(64N) |
    //                     agg0(64N) | H2(48N) | Wp0(f16) | Wp1(f16)
    size_t need = (size_t)N * (4 + 64 + 64 + 48) * sizeof(float) + (4096 + 3072) * sizeof(_Float16);
    if (ws_size < need) return;

    float* ws       = (float*)d_ws;
    float* deg_out  = ws;
    float* deg_in   = ws + (size_t)N;
    float* s_out    = ws + (size_t)2 * N;
    float* s_in     = ws + (size_t)3 * N;
    float* H0       = ws + (size_t)4 * N;            // reused as X1 after layer-0 agg
    float* agg0     = H0 + (size_t)64 * N;
    float* H2       = agg0 + (size_t)64 * N;         // stride 48 (padded 40)
    _Float16* Wp0   = (_Float16*)(H2 + (size_t)48 * N);
    _Float16* Wp1   = Wp0 + 4096;

    // Zero accumulators (graph-capture-safe async memsets)
    hipMemsetAsync(deg_out, 0, (size_t)2 * N * sizeof(float), stream);
    hipMemsetAsync(agg0,    0, (size_t)64 * N * sizeof(float), stream);
    hipMemsetAsync(out,     0, (size_t)40 * N * sizeof(float), stream);

    // Degrees + normalization scales
    gcn_degrees<<<(E + 255) / 256, 256, 0, stream>>>(src, dst, deg_out, deg_in, E);
    gcn_rsqrt<<<(N + 255) / 256, 256, 0, stream>>>(deg_out, deg_in, s_out, s_in, N);

    // Pack weights into WMMA B-layout (f16)
    gcn_packW<<<16, 256, 0, stream>>>(W0, Wp0, 64, 4 * 2 * 32 * 16);  // 64 cols -> 4 tiles
    gcn_packW<<<12, 256, 0, stream>>>(W1, Wp1, 40, 3 * 2 * 32 * 16);  // 40 cols -> 3 tiles (padded)

    const int tiles = (N + 15) / 16;
    const int gblks = (tiles + 3) / 4;   // 4 waves (128 threads) per block

    // Layer 0: H0 = (X * s_out) @ W0
    gcn_gemm_wmma<4><<<gblks, 128, 0, stream>>>(in_feat, s_out, Wp0, H0, N, 64);
    // agg0[dst] += H0[src]
    gcn_scatter64<<<(E * 4 + 255) / 256, 256, 0, stream>>>(src, dst, H0, agg0, E);
    // X1 = relu(agg0 * s_in + b0) * s_out    (overwrites H0)
    gcn_relu_scale<<<(N * 16 + 255) / 256, 256, 0, stream>>>(agg0, s_in, s_out, b0, H0, N);

    // Layer 1: H2 = X1 @ W1   (padded to 48 cols; pad cols are exact zeros)
    gcn_gemm_wmma<3><<<gblks, 128, 0, stream>>>(H0, nullptr, Wp1, H2, N, 48);
    // out[dst] += H2[src] (first 40 cols)
    gcn_scatter40<<<(E * 5 + 255) / 256, 256, 0, stream>>>(src, dst, H2, out, E);
    // out = out * s_in + b1
    gcn_final<<<(N * 10 + 255) / 256, 256, 0, stream>>>(out, s_in, b1, N);
}
